// TransformerLM_64914135712372
// MI455X (gfx1250) — compile-verified, matrix operand order checked by reading, not run on hardware
//
#include <hip/hip_runtime.h>
#include <cstdint>
#include <cstddef>

// ---------------- problem constants ----------------
#define Vv   32000
#define Dm   1024
#define Hh   16
#define Ll   8
#define Bb   2
#define Sdim 1024
#define DFd  4096
#define BS   (Bb * Sdim)     // 2048
#define DK3  (3 * Dm)        // 3072

// ---------------- WMMA types ----------------
typedef __attribute__((ext_vector_type(16))) __bf16 v16bf;
typedef __attribute__((ext_vector_type(8)))  float  v8f;

union FragBf {
    v16bf    v;
    uint32_t u[8];
    uint16_t h[16];
};

__device__ __forceinline__ uint16_t f2bf(float f) {
    uint32_t u = __builtin_bit_cast(uint32_t, f);
    uint32_t r = u + 0x7FFFu + ((u >> 16) & 1u);   // round-to-nearest-even
    return (uint16_t)(r >> 16);
}

// K-offset pattern of the 16-bit 16x32 A-layout (and transposed B-layout):
// lanes 0-15: VGPR0-3 -> K {0,2,4,6}+, VGPR4-7 -> K {16..22}+ ; hi lanes +8.
__device__ __forceinline__ int joff_of(int v, int hi) {
    int b = hi * 8;
    return (v < 4) ? (b + 2 * v) : (16 + b + 2 * (v - 4));
}

__device__ __forceinline__ v8f wmma_bf16(const FragBf& a, const FragBf& b, v8f c) {
    return __builtin_amdgcn_wmma_f32_16x16x32_bf16(
        /*neg_a=*/false, a.v, /*neg_b=*/false, b.v,
        /*c_mod=*/(short)0, c, /*reuse_a=*/false, /*reuse_b=*/false);
}

// Async global->LDS copy of 16 bytes per lane (tracked by ASYNCcnt).
// LDS destination operand is the raw LDS byte offset = low 32 bits of the
// generic shared-memory address (LDS aperture: addr[31:0] is the LDS offset).
__device__ __forceinline__ void async_copy_b128(uint32_t lds_off, const void* gsrc) {
    asm volatile("global_load_async_to_lds_b128 %0, %1, off"
                 :: "v"(lds_off), "v"(gsrc)
                 : "memory");
}
__device__ __forceinline__ void wait_async0() {
    asm volatile("s_wait_asynccnt 0x0" ::: "memory");
}

// ---------------- embedding + sinusoidal PE ----------------
__global__ void embed_kernel(const int* __restrict__ x,
                             const float* __restrict__ tok_emb,
                             float* __restrict__ h) {
    int idx = blockIdx.x * blockDim.x + threadIdx.x;
    if (idx >= BS * Dm) return;
    int bs = idx / Dm;
    int d  = idx - bs * Dm;
    int s  = bs & (Sdim - 1);
    float freq = expf(-(float)(d & ~1) * (logf(10000.0f) / (float)Dm));
    float ang  = (float)s * freq;
    float pe   = (d & 1) ? cosf(ang) : sinf(ang);
    h[idx] = tok_emb[(size_t)x[bs] * Dm + d] + pe;
}

// ---------------- LayerNorm (fp32 in -> bf16 out) ----------------
__global__ __launch_bounds__(256)
void ln_kernel(const float* __restrict__ xin,
               const float* __restrict__ g,
               const float* __restrict__ bta,
               uint16_t* __restrict__ out) {
    __shared__ float red[256];
    const int row = blockIdx.x;
    const int t   = threadIdx.x;
    const float* xr = xin + (size_t)row * Dm;

    float xv[4];
    float s = 0.f;
    for (int i = 0; i < 4; ++i) { xv[i] = xr[t + 256 * i]; s += xv[i]; }
    red[t] = s; __syncthreads();
    for (int off = 128; off > 0; off >>= 1) {
        if (t < off) red[t] += red[t + off];
        __syncthreads();
    }
    float mu = red[0] * (1.f / Dm);
    __syncthreads();

    float vs = 0.f;
    for (int i = 0; i < 4; ++i) { float d = xv[i] - mu; vs += d * d; }
    red[t] = vs; __syncthreads();
    for (int off = 128; off > 0; off >>= 1) {
        if (t < off) red[t] += red[t + off];
        __syncthreads();
    }
    float rstd = rsqrtf(red[0] * (1.f / Dm) + 1e-5f);

    uint16_t* orow = out + (size_t)row * Dm;
    for (int i = 0; i < 4; ++i) {
        int idx = t + 256 * i;
        orow[idx] = f2bf((xv[i] - mu) * rstd * g[idx] + bta[idx]);
    }
}

// ---------------- generic WMMA GEMM ----------------
// C(MxN) = epilogue( A_bf16(MxK) @ W_fp32(KxN) + bias? ) (+residual?) -> fp32 or bf16
// Block: 256 threads = 8 waves (4 M-groups x 2 N-groups), tile 128x128, BK=32.
// A tile staged via ASYNCcnt-tracked global_load_async_to_lds_b128 (DMA runs
// while the fp32->bf16 W-tile conversion executes on the VALU).
template<bool GELU_, bool OUT_BF16>
__global__ __launch_bounds__(256)
void gemm_kernel(const uint16_t* __restrict__ A,
                 const float* __restrict__ W,
                 const float* __restrict__ bias,
                 const float* __restrict__ residual,
                 float* __restrict__ outF,
                 uint16_t* __restrict__ outB,
                 int M, int N, int K) {
    __shared__ uint16_t lds_a[128 * 32];   // [m][k] row-major
    __shared__ uint16_t lds_b[128 * 32];   // [n][k] (transposed B)

    const int t    = threadIdx.x;
    const int lane = t & 31, li = lane & 15, hi = lane >> 4;
    const int wid  = t >> 5;
    const int wm   = wid & 3;        // 0..3 -> 32-row group
    const int wn   = wid >> 2;       // 0..1 -> 64-col group
    const int tileM = blockIdx.y * 128;
    const int tileN = blockIdx.x * 128;

    v8f acc[2][4];
    for (int mi = 0; mi < 2; ++mi)
        for (int ni = 0; ni < 4; ++ni)
            for (int r = 0; r < 8; ++r) acc[mi][ni][r] = 0.f;

    const int nK = K >> 5;
    for (int ko = 0; ko < nK; ++ko) {
        // 1) kick off async DMA of the A tile: 128 rows x 32 bf16, 16B/lane/pass
        {
            int c0 = (t & 3) * 8;
            for (int i = 0; i < 2; ++i) {
                int r = i * 64 + (t >> 2);
                const void* gsrc = (const void*)(A + (size_t)(tileM + r) * K + ko * 32 + c0);
                uint32_t ldst = (uint32_t)(uintptr_t)&lds_a[r * 32 + c0];
                async_copy_b128(ldst, gsrc);
            }
        }
        // 2) overlap: stage W tile 32(k) x 128(n) fp32 -> bf16, transposed [n][k]
        for (int i = 0; i < 16; ++i) {
            int e = t + (i << 8);
            int k = e >> 7, n = e & 127;
            float w = W[(size_t)(ko * 32 + k) * N + tileN + n];
            lds_b[n * 32 + k] = f2bf(w);
        }
        // 3) prefetch next W tile into cache (global_prefetch_b8), 1 line/thread
        if (ko + 1 < nK) {
            const float* pf = W + (size_t)((ko + 1) * 32 + (t >> 3)) * N
                                + tileN + (t & 7) * 16;
            __builtin_prefetch(pf, 0, 1);
        }
        wait_async0();
        __syncthreads();

        FragBf af[2], bfq[4];
        for (int mi = 0; mi < 2; ++mi) {
            int row = wm * 32 + mi * 16 + li;
            for (int v = 0; v < 8; ++v)
                af[mi].u[v] = *reinterpret_cast<const uint32_t*>(
                    &lds_a[row * 32 + joff_of(v, hi)]);
        }
        for (int ni = 0; ni < 4; ++ni) {
            int col = wn * 64 + ni * 16 + li;
            for (int v = 0; v < 8; ++v)
                bfq[ni].u[v] = *reinterpret_cast<const uint32_t*>(
                    &lds_b[col * 32 + joff_of(v, hi)]);
        }
        for (int mi = 0; mi < 2; ++mi)
            for (int ni = 0; ni < 4; ++ni)
                acc[mi][ni] = wmma_bf16(af[mi], bfq[ni], acc[mi][ni]);
        __syncthreads();
    }

    // epilogue (C layout: lane li -> N, VGPR r -> M row r + hi*8)
    for (int mi = 0; mi < 2; ++mi) {
        for (int ni = 0; ni < 4; ++ni) {
            int gcol = tileN + wn * 64 + ni * 16 + li;
            float bv = bias ? bias[gcol] : 0.f;
            for (int r = 0; r < 8; ++r) {
                int grow = tileM + wm * 32 + mi * 16 + r + hi * 8;
                float val = acc[mi][ni][r] + bv;
                if (GELU_) val = 0.5f * val * (1.f + erff(val * 0.70710678118f));
                size_t o = (size_t)grow * N + gcol;
                if (residual) val += residual[o];
                if (OUT_BF16) outB[o] = f2bf(val);
                else          outF[o] = val;
            }
        }
    }
}

// ---------------- flash attention (causal, per-head) ----------------
// qkv: bf16 [B*S, 3*D], col = c*1024 + head*64 + dk. One wave = 16 q rows.
__global__ __launch_bounds__(128)
void attn_kernel(const uint16_t* __restrict__ qkv,
                 uint16_t* __restrict__ attn_out) {
    __shared__ uint16_t lds_p[4][16 * 32];   // per-wave P staging (C->A relayout)

    const int t    = threadIdx.x;
    const int lane = t & 31, li = lane & 15, hi = lane >> 4;
    const int wid  = t >> 5;
    const int gt   = blockIdx.x * 4 + wid;   // 0 .. B*H*(S/16)-1 = 2047
    const int b    = gt >> 10;
    const int rem  = gt & 1023;
    const int head = rem >> 6;
    const int qt   = rem & 63;
    const int q0   = qt << 4;

    // Q as two A-fragments (16 rows x dk 0..31 / 32..63), packed-K b32 loads
    FragBf qa[2];
    {
        const uint32_t* qrow = reinterpret_cast<const uint32_t*>(
            qkv + (size_t)(b * Sdim + q0 + li) * DK3 + head * 64);
        for (int kk = 0; kk < 2; ++kk)
            for (int v = 0; v < 8; ++v)
                qa[kk].u[v] = qrow[(kk * 32 + joff_of(v, hi)) >> 1];
    }

    v8f o[4];
    for (int nt = 0; nt < 4; ++nt)
        for (int r = 0; r < 8; ++r) o[nt][r] = 0.f;
    float mrun[8], lrun[8];
    for (int r = 0; r < 8; ++r) { mrun[r] = -1e30f; lrun[r] = 0.f; }

    const int ktmax = (q0 + 15) >> 5;
    for (int kt = 0; kt <= ktmax; ++kt) {
        const int k0 = kt << 5;
        float sv[2][8];

        // scores: two 16x16 tiles, contraction over dk=64 (2 WMMA steps each)
        for (int st = 0; st < 2; ++st) {
            FragBf kb[2];
            const uint32_t* krow = reinterpret_cast<const uint32_t*>(
                qkv + (size_t)(b * Sdim + k0 + st * 16 + li) * DK3 + 1024 + head * 64);
            for (int kk = 0; kk < 2; ++kk)
                for (int v = 0; v < 8; ++v)
                    kb[kk].u[v] = krow[(kk * 32 + joff_of(v, hi)) >> 1];
            v8f sc;
            for (int r = 0; r < 8; ++r) sc[r] = 0.f;
            sc = wmma_bf16(qa[0], kb[0], sc);
            sc = wmma_bf16(qa[1], kb[1], sc);
            const int col = k0 + st * 16 + li;
            for (int r = 0; r < 8; ++r) {
                int row = q0 + r + hi * 8;
                float s = sc[r] * 0.125f;      // 1/sqrt(DK)
                sv[st][r] = (col > row) ? -1e30f : s;
            }
        }

        // online softmax: row reductions across each 16-lane half
        float pv[2][8], fscale[8];
        for (int r = 0; r < 8; ++r) {
            float rm = fmaxf(sv[0][r], sv[1][r]);
            for (int m2 = 1; m2 < 16; m2 <<= 1) rm = fmaxf(rm, __shfl_xor(rm, m2, 32));
            float nm = fmaxf(mrun[r], rm);
            float p0 = expf(sv[0][r] - nm);
            float p1 = expf(sv[1][r] - nm);
            float rs = p0 + p1;
            for (int m2 = 1; m2 < 16; m2 <<= 1) rs += __shfl_xor(rs, m2, 32);
            float f  = expf(mrun[r] - nm);
            lrun[r]  = lrun[r] * f + rs;
            mrun[r]  = nm;
            fscale[r] = f;
            pv[0][r] = p0; pv[1][r] = p1;
        }
        for (int nt = 0; nt < 4; ++nt)
            for (int r = 0; r < 8; ++r) o[nt][r] *= fscale[r];

        // P: C-layout -> LDS -> A-layout (same wave; just order DS ops)
        for (int st = 0; st < 2; ++st)
            for (int r = 0; r < 8; ++r)
                lds_p[wid][(r + hi * 8) * 32 + st * 16 + li] = f2bf(pv[st][r]);
        asm volatile("s_wait_dscnt 0x0" ::: "memory");
        FragBf pa;
        for (int v = 0; v < 8; ++v)
            pa.u[v] = *reinterpret_cast<const uint32_t*>(
                &lds_p[wid][li * 32 + joff_of(v, hi)]);

        // P(16x32) @ V(32x64): four N-tiles; V B-frag needs two rows per VGPR
        for (int nt = 0; nt < 4; ++nt) {
            FragBf vb;
            const size_t vcol = (size_t)2048 + head * 64 + nt * 16 + li;
            for (int v = 0; v < 8; ++v) {
                int j0 = joff_of(v, hi);
                uint16_t x0 = qkv[(size_t)(b * Sdim + k0 + j0)     * DK3 + vcol];
                uint16_t x1 = qkv[(size_t)(b * Sdim + k0 + j0 + 1) * DK3 + vcol];
                vb.u[v] = (uint32_t)x0 | ((uint32_t)x1 << 16);
            }
            o[nt] = wmma_bf16(pa, vb, o[nt]);
        }
    }

    for (int nt = 0; nt < 4; ++nt)
        for (int r = 0; r < 8; ++r) {
            int row = q0 + r + hi * 8;
            float val = o[nt][r] / lrun[r];
            attn_out[(size_t)(b * Sdim + row) * Dm + head * 64 + nt * 16 + li] = f2bf(val);
        }
}

// ---------------- host orchestration ----------------
extern "C" void kernel_launch(void* const* d_in, const int* in_sizes, int n_in,
                              void* d_out, int out_size, void* d_ws, size_t ws_size,
                              hipStream_t stream) {
    (void)in_sizes; (void)n_in; (void)out_size; (void)ws_size;

    const int*   x    = (const int*)  d_in[0];
    const float* tok  = (const float*)d_in[1];
    const float* Wqkv = (const float*)d_in[2];
    const float* Wout = (const float*)d_in[3];
    const float* ln1g = (const float*)d_in[4];
    const float* ln1b = (const float*)d_in[5];
    const float* W1   = (const float*)d_in[6];
    const float* b1   = (const float*)d_in[7];
    const float* W2   = (const float*)d_in[8];
    const float* b2   = (const float*)d_in[9];
    const float* ln2g = (const float*)d_in[10];
    const float* ln2b = (const float*)d_in[11];
    const float* lnfg = (const float*)d_in[12];
    const float* lnfb = (const float*)d_in[13];
    const float* Wh   = (const float*)d_in[14];
    const float* bh   = (const float*)d_in[15];

    char* ws = (char*)d_ws;
    size_t off = 0;
    auto alloc = [&](size_t bytes) {
        void* p = ws + off;
        off += (bytes + 255) & ~(size_t)255;
        return p;
    };
    float*    h    = (float*)   alloc((size_t)BS * Dm  * 4);
    uint16_t* xln  = (uint16_t*)alloc((size_t)BS * Dm  * 2);
    uint16_t* qkvb = (uint16_t*)alloc((size_t)BS * DK3 * 2);
    uint16_t* aout = (uint16_t*)alloc((size_t)BS * Dm  * 2);
    uint16_t* mid  = (uint16_t*)alloc((size_t)BS * DFd * 2);

    embed_kernel<<<(BS * Dm + 255) / 256, 256, 0, stream>>>(x, tok, h);

    for (int l = 0; l < Ll; ++l) {
        ln_kernel<<<BS, 256, 0, stream>>>(h, ln1g + l * Dm, ln1b + l * Dm, xln);

        gemm_kernel<false, true><<<dim3(DK3 / 128, BS / 128), 256, 0, stream>>>(
            xln, Wqkv + (size_t)l * Dm * DK3, nullptr, nullptr, nullptr, qkvb,
            BS, DK3, Dm);

        attn_kernel<<<(Bb * Hh * (Sdim / 16)) / 4, 128, 0, stream>>>(qkvb, aout);

        gemm_kernel<false, false><<<dim3(Dm / 128, BS / 128), 256, 0, stream>>>(
            aout, Wout + (size_t)l * Dm * Dm, nullptr, h, h, nullptr,
            BS, Dm, Dm);

        ln_kernel<<<BS, 256, 0, stream>>>(h, ln2g + l * Dm, ln2b + l * Dm, xln);

        gemm_kernel<true, true><<<dim3(DFd / 128, BS / 128), 256, 0, stream>>>(
            xln, W1 + (size_t)l * Dm * DFd, b1 + l * DFd, nullptr, nullptr, mid,
            BS, DFd, Dm);

        gemm_kernel<false, false><<<dim3(Dm / 128, BS / 128), 256, 0, stream>>>(
            mid, W2 + (size_t)l * DFd * Dm, b2 + l * Dm, h, h, nullptr,
            BS, Dm, DFd);
    }

    ln_kernel<<<BS, 256, 0, stream>>>(h, lnfg, lnfb, xln);

    gemm_kernel<false, false><<<dim3(Vv / 128, BS / 128), 256, 0, stream>>>(
        xln, Wh, bh, nullptr, (float*)d_out, nullptr,
        BS, Vv, Dm);
}
